// AGCRNLayer_56667798504187
// MI455X (gfx1250) — matrix operand order repaired and checked by reading, not compiled, not run on hardware
//
#include <hip/hip_runtime.h>
#include <math.h>

typedef __bf16 bf16;
typedef __attribute__((ext_vector_type(16))) __bf16 v16bf;
typedef __attribute__((ext_vector_type(8)))  float  v8f;

// ---------------- problem constants ----------------
constexpr int N_NODES = 2000;
constexpr int NP      = 2048;              // padded node count (mult of 64)
constexpr int DIN     = 2;
constexpr int DOUT    = 64;
constexpr int EDIM    = 16;
constexpr int KCH     = 3;
constexpr int BATCH   = 64;
constexpr int FEAT    = DIN + DOUT;        // 66
constexpr int CCOLS   = BATCH * FEAT;      // 4224  (mult of 128)
constexpr int KFLAT   = KCH * FEAT;        // 198
constexpr int KPAD    = 224;               // 7 * 32

// ---------------- WMMA helpers ----------------
__device__ __forceinline__ v8f wmma_bf16(v16bf a, v16bf b, v8f c) {
  return __builtin_amdgcn_wmma_f32_16x16x32_bf16(false, a, false, b, (short)0, c,
                                                 false, false);
}

// A-fragment: 16x32 bf16 tile at src (element (m,k) at src[m*ld+k]).
// ISA layout: lanes 0-15 -> M=lane, K base 0; lanes 16-31 -> M=lane-16, K base 8.
// VGPR v (pair 2v,2v+1): K = kbase + (v>=4 ? 16:0) + 2*(v%4) + h.
__device__ __forceinline__ v16bf load_a_frag(const bf16* src, int ld, int lane) {
  int m  = lane & 15;
  int kb = (lane >> 4) << 3;            // 0 or 8
  const bf16* row = src + m * ld + kb;
  v16bf a;
#pragma unroll
  for (int v = 0; v < 8; ++v) {
    int k = ((v >> 2) << 4) + ((v & 3) << 1);  // 0,2,4,6,16,18,20,22
    a[2 * v]     = row[k];
    a[2 * v + 1] = row[k + 1];
  }
  return a;
}

// B-fragment: 32x16 bf16 tile at src (element (k,n) at src[k*ld+n]).
// lane -> N=lane%16, vector element j -> K in same order as A fragment.
__device__ __forceinline__ v16bf load_b_frag(const bf16* src, int ld, int lane) {
  const bf16* col = src + (lane & 15) + ((lane >> 4) << 3) * ld;
  v16bf b;
#pragma unroll
  for (int j = 0; j < 16; ++j) b[j] = col[(((j >> 3) << 4) + (j & 7)) * ld];
  return b;
}

// ---------------- CDNA5 async global->LDS copy (ASYNCcnt-tracked) ----------------
// Generic LDS pointer's low 32 bits are the LDS offset (AS3 -> generic keeps it).
__device__ __forceinline__ void async_cp16(void* lds, const void* g) {
  unsigned l = (unsigned)(unsigned long long)lds;
  unsigned long long ga = (unsigned long long)g;
  asm volatile("global_load_async_to_lds_b128 %0, %1, off"
               :: "v"(l), "v"(ga) : "memory");
}
__device__ __forceinline__ void wait_async0() {
  asm volatile("s_wait_asynccnt 0x0" ::: "memory");
}

// ---------------- kernel 1: adjacency ----------------
// A[m,n] = softmax_n(relu(emb[m].emb[n])), bf16, padded NPxNP (pad = 0)
__global__ __launch_bounds__(256) void adjacency_kernel(
    const float* __restrict__ emb, bf16* __restrict__ A) {
  int m   = blockIdx.x;
  int tid = threadIdx.x;
  if (m >= N_NODES) {
    for (int n2 = tid; n2 < NP; n2 += 256) A[(size_t)m * NP + n2] = (bf16)0.f;
    return;
  }
  __shared__ float semb[EDIM];
  __shared__ float slog[NP];
  __shared__ float red[256];
  if (tid < EDIM) semb[tid] = emb[(size_t)m * EDIM + tid];
  __syncthreads();
  float lmax = -1e30f;
  for (int n2 = tid; n2 < NP; n2 += 256) {
    float v = 0.f;
    if (n2 < N_NODES) {
      const float* er = emb + (size_t)n2 * EDIM;
#pragma unroll
      for (int e = 0; e < EDIM; ++e) v += semb[e] * er[e];
      v = fmaxf(v, 0.f);
      lmax = fmaxf(lmax, v);
    }
    slog[n2] = v;
  }
  red[tid] = lmax;
  __syncthreads();
  for (int s = 128; s > 0; s >>= 1) {
    if (tid < s) red[tid] = fmaxf(red[tid], red[tid + s]);
    __syncthreads();
  }
  float rmax = red[0];
  __syncthreads();
  float lsum = 0.f;
  for (int n2 = tid; n2 < N_NODES; n2 += 256) {
    float e = __expf(slog[n2] - rmax);
    slog[n2] = e;
    lsum += e;
  }
  red[tid] = lsum;
  __syncthreads();
  for (int s = 128; s > 0; s >>= 1) {
    if (tid < s) red[tid] += red[tid + s];
    __syncthreads();
  }
  float inv = 1.f / red[0];
  __syncthreads();
  for (int n2 = tid; n2 < NP; n2 += 256) {
    float v = (n2 < N_NODES) ? slog[n2] * inv : 0.f;
    A[(size_t)m * NP + n2] = (bf16)v;
  }
}

// ---------------- kernel 2: pack xs -> Y0 (node-major bf16), zero pads ----------------
__global__ __launch_bounds__(256) void pack_kernel(
    const float* __restrict__ x, const float* __restrict__ state,
    bf16* __restrict__ Y0, bf16* __restrict__ C0) {
  size_t idx = (size_t)blockIdx.x * 256 + threadIdx.x;
  if (idx >= (size_t)NP * CCOLS) return;
  int n = (int)(idx / CCOLS);
  int c = (int)(idx % CCOLS);
  int b = c / FEAT, i = c - b * FEAT;
  if (n < N_NODES) {
    float v = (i < DIN) ? x[((size_t)b * N_NODES + n) * DIN + i]
                        : state[((size_t)b * N_NODES + n) * DOUT + (i - DIN)];
    Y0[idx] = (bf16)v;
  } else {
    Y0[idx] = (bf16)0.f;
    C0[idx] = (bf16)0.f;   // zero pad rows of candidate features too
  }
}

// ---------------- kernel 3: out = alpha*A@X (+ beta*Csub), all bf16 node-major ----------------
// A: [NP,NP], X/out/Csub: [NP,CCOLS]. Block tile 64x128, K step 32, 8 waves,
// double-buffered LDS fed by async global->LDS copies.
__global__ __launch_bounds__(256) void gemm_prop(
    const bf16* __restrict__ A, const bf16* __restrict__ X,
    bf16* __restrict__ out, const bf16* __restrict__ Csub,
    float alpha, float beta) {
  __shared__ alignas(16) bf16 sA[2][64][40];
  __shared__ alignas(16) bf16 sB[2][32][136];
  int tid = threadIdx.x;
  int m0 = blockIdx.y * 64;
  int n0 = blockIdx.x * 128;
  int w = tid >> 5, lane = tid & 31;
  int tr = w >> 1;               // output row tile 0..3
  int cg = (w & 1) * 4;          // output col tiles cg..cg+3 (of 8)
  v8f acc[4] = {};
  int ra = tid >> 2, sega = (tid & 3) * 8;   // 64 rows x 4 segs of 8 bf16
  int rb = tid >> 3, segb = (tid & 7) * 8;   // 32 rows x 8 segs of 8 bf16 (x2 halves)
  const bf16* gA = &A[(size_t)(m0 + ra) * NP + sega];
  const bf16* gX = &X[(size_t)rb * CCOLS + n0 + segb];

  auto load_tiles = [&](int kk0, int buf) {
    async_cp16(&sA[buf][ra][sega],      gA + kk0);
    async_cp16(&sB[buf][rb][segb],      gX + (size_t)kk0 * CCOLS);
    async_cp16(&sB[buf][rb][segb + 64], gX + (size_t)kk0 * CCOLS + 64);
  };

  load_tiles(0, 0);
  int cur = 0;
  for (int kk0 = 0; kk0 < NP; kk0 += 32) {
    wait_async0();          // our async writes into sA/sB[cur] have landed
    __syncthreads();        // visible to all; prior reads of sA/sB[cur^1] done
    if (kk0 + 32 < NP) load_tiles(kk0 + 32, cur ^ 1);
    if (kk0 + 64 < NP) {    // warm L2 two steps ahead
      __builtin_prefetch(gA + kk0 + 64, 0, 1);
      __builtin_prefetch(gX + (size_t)(kk0 + 64) * CCOLS, 0, 1);
    }
    v16bf a = load_a_frag(&sA[cur][tr * 16][0], 40, lane);
#pragma unroll
    for (int c2 = 0; c2 < 4; ++c2) {
      v16bf bb = load_b_frag(&sB[cur][0][(cg + c2) * 16], 136, lane);
      acc[c2] = wmma_bf16(a, bb, acc[c2]);
    }
    cur ^= 1;
  }
  int mb = m0 + tr * 16 + ((lane >> 4) << 3);
#pragma unroll
  for (int c2 = 0; c2 < 4; ++c2) {
    int n = n0 + (cg + c2) * 16 + (lane & 15);
#pragma unroll
    for (int rr = 0; rr < 8; ++rr) {
      size_t i0 = (size_t)(mb + rr) * CCOLS + n;
      float v = alpha * acc[c2][rr];
      if (Csub) v += beta * (float)Csub[i0];
      out[i0] = (bf16)v;
    }
  }
}

// ---------------- kernel 4: reset/update gates per node ----------------
// z_r[b,o2] = sigmoid( sum_{k,i} Yk[n,b,i] * (emb[n].W_reset)[k,i,o2] + (emb[n].b_reset)[o2] )
// writes zbuf[b,n,o] (o2<64) and C0[n, b*66+2+o] = r*state (o2>=64); also x part of C0.
__global__ __launch_bounds__(256) void gates_kernel(
    const bf16* __restrict__ Y0, const bf16* __restrict__ Y1, const bf16* __restrict__ Y2,
    const float* __restrict__ emb, const float* __restrict__ Wr, const float* __restrict__ br,
    const float* __restrict__ x, const float* __restrict__ state,
    float* __restrict__ zbuf, bf16* __restrict__ C0) {
  int n   = blockIdx.x;
  int tid = threadIdx.x;
  __shared__ alignas(16) bf16 sX[64][232];   // 64 batch rows x 224 K (padded)
  __shared__ alignas(16) bf16 sW[32][136];   // one K-chunk of node weights
  __shared__ float semb[EDIM];
  __shared__ float sbias[128];
  if (tid < EDIM) semb[tid] = emb[(size_t)n * EDIM + tid];
  __syncthreads();
  if (tid < 128) {
    float s = 0.f;
#pragma unroll
    for (int e = 0; e < EDIM; ++e) s += semb[e] * br[e * 128 + tid];
    sbias[tid] = s;
  }
  for (int idx = tid; idx < 64 * KPAD; idx += 256) {
    int b = idx / KPAD, kk = idx - b * KPAD;
    bf16 v = (bf16)0.f;
    if (kk < KFLAT) {
      int k = kk / FEAT, i = kk - k * FEAT;
      const bf16* Yk = (k == 0) ? Y0 : (k == 1) ? Y1 : Y2;
      v = Yk[(size_t)n * CCOLS + b * FEAT + i];
    }
    sX[b][kk] = v;
  }
  if (tid < BATCH * DIN) {  // x part of candidate features
    int b = tid >> 1, i = tid & 1;
    C0[(size_t)n * CCOLS + b * FEAT + i] =
        (bf16)x[((size_t)b * N_NODES + n) * DIN + i];
  }
  __syncthreads();
  int w = tid >> 5, lane = tid & 31;
  int tr = w >> 1;          // batch tile row 0..3
  int cg = (w & 1) * 4;     // col tile group base (of 8 col tiles)
  v8f acc[4] = {};
  for (int ck = 0; ck < 7; ++ck) {
    for (int idx = tid; idx < 32 * 128; idx += 256) {
      int rr = idx >> 7, o = idx & 127;
      int kk = ck * 32 + rr;
      float v = 0.f;
      if (kk < KFLAT) {
        int k = kk / FEAT, i = kk - k * FEAT;
        const float* wp = Wr + ((size_t)k * FEAT + i) * 128 + o;
#pragma unroll
        for (int e = 0; e < EDIM; ++e) v += semb[e] * wp[(size_t)e * KFLAT * 128];
      }
      sW[rr][o] = (bf16)v;
    }
    __syncthreads();
    v16bf a = load_a_frag(&sX[tr * 16][ck * 32], 232, lane);
#pragma unroll
    for (int c2 = 0; c2 < 4; ++c2) {
      v16bf bb = load_b_frag(&sW[0][(cg + c2) * 16], 136, lane);
      acc[c2] = wmma_bf16(a, bb, acc[c2]);
    }
    __syncthreads();
  }
#pragma unroll
  for (int c2 = 0; c2 < 4; ++c2) {
    int o2 = (cg + c2) * 16 + (lane & 15);
    float bias = sbias[o2];
#pragma unroll
    for (int rr = 0; rr < 8; ++rr) {
      int b = tr * 16 + rr + ((lane >> 4) << 3);
      float g = 1.f / (1.f + __expf(-(acc[c2][rr] + bias)));
      size_t bi = (size_t)b * N_NODES + n;
      if (o2 < DOUT) {
        zbuf[bi * DOUT + o2] = g;
      } else {
        int o = o2 - DOUT;
        float st = state[bi * DOUT + o];
        C0[(size_t)n * CCOLS + b * FEAT + DIN + o] = (bf16)(g * st);
      }
    }
  }
}

// ---------------- kernel 5: candidate + combine ----------------
__global__ __launch_bounds__(256) void cand_kernel(
    const bf16* __restrict__ C0, const bf16* __restrict__ C1, const bf16* __restrict__ C2,
    const float* __restrict__ emb, const float* __restrict__ Wu, const float* __restrict__ bu,
    const float* __restrict__ state, const float* __restrict__ zbuf,
    float* __restrict__ out) {
  int n   = blockIdx.x;
  int tid = threadIdx.x;
  __shared__ alignas(16) bf16 sX[64][232];
  __shared__ alignas(16) bf16 sW[32][72];
  __shared__ float semb[EDIM];
  __shared__ float sbias[64];
  if (tid < EDIM) semb[tid] = emb[(size_t)n * EDIM + tid];
  __syncthreads();
  if (tid < 64) {
    float s = 0.f;
#pragma unroll
    for (int e = 0; e < EDIM; ++e) s += semb[e] * bu[e * 64 + tid];
    sbias[tid] = s;
  }
  for (int idx = tid; idx < 64 * KPAD; idx += 256) {
    int b = idx / KPAD, kk = idx - b * KPAD;
    bf16 v = (bf16)0.f;
    if (kk < KFLAT) {
      int k = kk / FEAT, i = kk - k * FEAT;
      const bf16* Ck = (k == 0) ? C0 : (k == 1) ? C1 : C2;
      v = Ck[(size_t)n * CCOLS + b * FEAT + i];
    }
    sX[b][kk] = v;
  }
  __syncthreads();
  int w = tid >> 5, lane = tid & 31;
  int tr = w >> 1;          // batch tile row 0..3
  int cg = (w & 1) * 2;     // col tile group (of 4 col tiles)
  v8f acc[2] = {};
  for (int ck = 0; ck < 7; ++ck) {
    for (int idx = tid; idx < 32 * 64; idx += 256) {
      int rr = idx >> 6, o = idx & 63;
      int kk = ck * 32 + rr;
      float v = 0.f;
      if (kk < KFLAT) {
        int k = kk / FEAT, i = kk - k * FEAT;
        const float* wp = Wu + ((size_t)k * FEAT + i) * 64 + o;
#pragma unroll
        for (int e = 0; e < EDIM; ++e) v += semb[e] * wp[(size_t)e * KFLAT * 64];
      }
      sW[rr][o] = (bf16)v;
    }
    __syncthreads();
    v16bf a = load_a_frag(&sX[tr * 16][ck * 32], 232, lane);
#pragma unroll
    for (int c2 = 0; c2 < 2; ++c2) {
      v16bf bb = load_b_frag(&sW[0][(cg + c2) * 16], 72, lane);
      acc[c2] = wmma_bf16(a, bb, acc[c2]);
    }
    __syncthreads();
  }
#pragma unroll
  for (int c2 = 0; c2 < 2; ++c2) {
    int o = (cg + c2) * 16 + (lane & 15);
    float bias = sbias[o];
#pragma unroll
    for (int rr = 0; rr < 8; ++rr) {
      int b = tr * 16 + rr + ((lane >> 4) << 3);
      float hc = tanhf(acc[c2][rr] + bias);
      size_t idx = ((size_t)b * N_NODES + n) * DOUT + o;
      float z = zbuf[idx];
      out[idx] = z * state[idx] + (1.f - z) * hc;
    }
  }
}

// ---------------- launch ----------------
extern "C" void kernel_launch(void* const* d_in, const int* in_sizes, int n_in,
                              void* d_out, int out_size, void* d_ws, size_t ws_size,
                              hipStream_t stream) {
  (void)in_sizes; (void)n_in; (void)out_size; (void)ws_size;
  const float* x     = (const float*)d_in[0];
  const float* state = (const float*)d_in[1];
  const float* emb   = (const float*)d_in[2];
  const float* Wr    = (const float*)d_in[3];
  const float* Wu    = (const float*)d_in[4];
  const float* br    = (const float*)d_in[5];
  const float* bu    = (const float*)d_in[6];
  float* out = (float*)d_out;

  char* ws = (char*)d_ws;
  const size_t szA = (size_t)NP * NP * sizeof(bf16);       //  8.4 MB
  const size_t szY = (size_t)NP * CCOLS * sizeof(bf16);    // 17.3 MB each
  bf16*  A    = (bf16*)(ws);
  bf16*  Y0   = (bf16*)(ws + szA);
  bf16*  Y1   = (bf16*)(ws + szA + szY);
  bf16*  Y2   = (bf16*)(ws + szA + 2 * szY);
  bf16*  C0   = (bf16*)(ws + szA + 3 * szY);
  float* zbuf = (float*)(ws + szA + 4 * szY);              // 32.8 MB
  bf16*  C1 = Y0;   // Y0 dead after gates_kernel
  bf16*  C2 = Y1;   // Y1 dead after gates_kernel

  adjacency_kernel<<<NP, 256, 0, stream>>>(emb, A);

  size_t tot = (size_t)NP * CCOLS;
  pack_kernel<<<(int)((tot + 255) / 256), 256, 0, stream>>>(x, state, Y0, C0);

  dim3 g(CCOLS / 128, NP / 64);
  gemm_prop<<<g, 256, 0, stream>>>(A, Y0, Y1, nullptr, 1.f, 0.f);   // Y1 = A@Y0
  gemm_prop<<<g, 256, 0, stream>>>(A, Y1, Y2, Y0, 2.f, -1.f);       // Y2 = 2A@Y1 - Y0

  gates_kernel<<<N_NODES, 256, 0, stream>>>(Y0, Y1, Y2, emb, Wr, br, x, state,
                                            zbuf, C0);

  gemm_prop<<<g, 256, 0, stream>>>(A, C0, C1, nullptr, 1.f, 0.f);   // C1 = A@C0
  gemm_prop<<<g, 256, 0, stream>>>(A, C1, C2, C0, 2.f, -1.f);       // C2 = 2A@C1 - C0

  cand_kernel<<<N_NODES, 256, 0, stream>>>(C0, C1, C2, emb, Wu, bu, state, zbuf,
                                           out);
}
